// GCN_37469294690813
// MI455X (gfx1250) — matrix operand order repaired and checked by reading, not compiled, not run on hardware
//
#include <hip/hip_runtime.h>
#include <cstdint>
#include <cstddef>

#define F_IN   128
#define HID    256
#define NGRAPH 256
#define NCLS   10

typedef _Float16 v4h  __attribute__((ext_vector_type(4)));
typedef _Float16 v8h  __attribute__((ext_vector_type(8)));
typedef _Float16 v16h __attribute__((ext_vector_type(16)));
typedef float    v8f  __attribute__((ext_vector_type(8)));

struct HalfPair { v8h lo, hi; };
static __device__ __forceinline__ v16h cat16(v8h lo, v8h hi) {
  HalfPair p; p.lo = lo; p.hi = hi;
  return __builtin_bit_cast(v16h, p);
}

static __device__ __forceinline__ void atomAddF(float* p, float v) {
  unsafeAtomicAdd(p, v);   // native global_atomic_add_f32
}

// ---------------- small utility kernels ----------------

__global__ void k_fill(float* __restrict__ p, float v, int n) {
  int i = blockIdx.x * blockDim.x + threadIdx.x;
  if (i < n) p[i] = v;
}

// degree: deg pre-filled with 1.0 (self-loop); add 1 per in-edge
__global__ void k_deg(const int* __restrict__ dst, float* __restrict__ deg, int E) {
  int e = blockIdx.x * blockDim.x + threadIdx.x;
  if (e < E) atomAddF(&deg[dst[e]], 1.0f);
}

// dinv = rsqrt(deg) in place (deg >= 1 always)
__global__ void k_rsqrt(float* __restrict__ p, int n) {
  int i = blockIdx.x * blockDim.x + threadIdx.x;
  if (i < n) p[i] = rsqrtf(p[i]);
}

__global__ void k_f32tof16(const float* __restrict__ in, _Float16* __restrict__ out, int n) {
  int i = blockIdx.x * blockDim.x + threadIdx.x;
  if (i < n) out[i] = (_Float16)in[i];
}

// W [K x HID] f32 row-major  ->  Wt [HID x K] f16 row-major (transposed)
__global__ void k_wprep(const float* __restrict__ W, _Float16* __restrict__ Wt, int K) {
  int i = blockIdx.x * blockDim.x + threadIdx.x;
  if (i < K * HID) {
    int n = i % HID;
    int k = i / HID;
    Wt[n * K + k] = (_Float16)W[i];
  }
}

// ---------------- WMMA GEMM: C[M x 256] = A[M x K](f16) @ Bt^T ----------------
// grid.x = M/16 row tiles; 128 threads = 4 waves; wave w owns col tiles [4w, 4w+4)
template<int K>
__global__ void __launch_bounds__(128)
k_gemm_wmma(const _Float16* __restrict__ A, const _Float16* __restrict__ Bt,
            float* __restrict__ C) {
  const int lane   = threadIdx.x & 31;
  const int wave   = threadIdx.x >> 5;
  const int laneLo = lane & 15;
  const int laneHi = lane >> 4;          // 0 or 1
  const int row0   = blockIdx.x << 4;

  // A fragment rows: M = row0 + laneLo
  const _Float16* __restrict__ arow = A + (size_t)(row0 + laneLo) * K;
  // B fragment: column n = tile*16 + laneLo ; lane group selects K half
  const _Float16* __restrict__ b0 = Bt + (size_t)((wave * 4 + 0) * 16 + laneLo) * K + laneHi * 16;
  const _Float16* __restrict__ b1 = Bt + (size_t)((wave * 4 + 1) * 16 + laneLo) * K + laneHi * 16;
  const _Float16* __restrict__ b2 = Bt + (size_t)((wave * 4 + 2) * 16 + laneLo) * K + laneHi * 16;
  const _Float16* __restrict__ b3 = Bt + (size_t)((wave * 4 + 3) * 16 + laneLo) * K + laneHi * 16;

  v8f acc0 = {}, acc1 = {}, acc2 = {}, acc3 = {};

  for (int k0 = 0; k0 < K; k0 += 32) {
    // A k order used here (B matches): halves = 16 consecutive K (lo lanes K0..15, hi lanes K16..31)
    v16h a = cat16(*(const v8h*)(arow + k0 + laneHi * 16),
                   *(const v8h*)(arow + k0 + laneHi * 16 + 8));
    // B k order used here (matches A): halves = 16 consecutive K (lo lanes K0..15, hi lanes K16..31)
    v16h vb0 = cat16(*(const v8h*)(b0 + k0), *(const v8h*)(b0 + k0 + 8));
    v16h vb1 = cat16(*(const v8h*)(b1 + k0), *(const v8h*)(b1 + k0 + 8));
    v16h vb2 = cat16(*(const v8h*)(b2 + k0), *(const v8h*)(b2 + k0 + 8));
    v16h vb3 = cat16(*(const v8h*)(b3 + k0), *(const v8h*)(b3 + k0 + 8));
    acc0 = __builtin_amdgcn_wmma_f32_16x16x32_f16(false, a, false, vb0, (short)0, acc0, false, false);
    acc1 = __builtin_amdgcn_wmma_f32_16x16x32_f16(false, a, false, vb1, (short)0, acc1, false, false);
    acc2 = __builtin_amdgcn_wmma_f32_16x16x32_f16(false, a, false, vb2, (short)0, acc2, false, false);
    acc3 = __builtin_amdgcn_wmma_f32_16x16x32_f16(false, a, false, vb3, (short)0, acc3, false, false);
  }

  // C/D layout: VGPR r, lanes 0-15 -> M=r, lanes 16-31 -> M=r+8 ; N = lane%16
  float* __restrict__ crow = C + (size_t)(row0 + 8 * laneHi) * HID + laneLo;
  const int c0 = (wave * 4 + 0) * 16, c1 = (wave * 4 + 1) * 16,
            c2 = (wave * 4 + 2) * 16, c3 = (wave * 4 + 3) * 16;
#pragma unroll
  for (int r = 0; r < 8; ++r) {
    float* cr = crow + (size_t)r * HID;
    cr[c0] = acc0[r]; cr[c1] = acc1[r]; cr[c2] = acc2[r]; cr[c3] = acc3[r];
  }
}

// ---------------- aggregation ----------------

// AGG[i,:] = H[i,:] * dinv[i]^2   (self-loop term; full overwrite, no zeroing pass)
__global__ void k_selfinit(const float* __restrict__ H, const float* __restrict__ dinv,
                           float* __restrict__ AGG, int n4) {
  int t = blockIdx.x * blockDim.x + threadIdx.x;
  if (t >= n4) return;
  int i = t >> 6;
  int c = (t & 63) << 2;
  float w = dinv[i]; w = w * w;
  float4 h = *(const float4*)(H + (size_t)i * HID + c);
  float4 o; o.x = h.x * w; o.y = h.y * w; o.z = h.z * w; o.w = h.w * w;
  *(float4*)(AGG + (size_t)i * HID + c) = o;
}

// per edge: AGG[dst,:] += H[src,:] * dinv[src]*dinv[dst]; 64 threads per edge
__global__ void k_edgeagg(const int* __restrict__ src, const int* __restrict__ dst,
                          const float* __restrict__ dinv, const float* __restrict__ H,
                          float* __restrict__ AGG, int E) {
  long long t = (long long)blockIdx.x * blockDim.x + threadIdx.x;
  int e = (int)(t >> 6);
  if (e >= E) return;
  int c = ((int)t & 63) << 2;
  // prefetch the gather row of a future edge to hide random-access latency
  int eN = e + 64;
  if (eN < E) {
    int sN = __builtin_nontemporal_load(&src[eN]);
    __builtin_prefetch(H + (size_t)sN * HID + c, 0, 0);
  }
  int s = src[e], d = dst[e];
  float w = dinv[s] * dinv[d];
  float4 h = *(const float4*)(H + (size_t)s * HID + c);
  float* o = AGG + (size_t)d * HID + c;
  atomAddF(o + 0, h.x * w);
  atomAddF(o + 1, h.y * w);
  atomAddF(o + 2, h.z * w);
  atomAddF(o + 3, h.w * w);
}

// next-layer input: f16( relu(AGG + b) )
__global__ void k_finalize(const float* __restrict__ AGG, const float* __restrict__ b,
                           _Float16* __restrict__ out, int n4) {
  int t = blockIdx.x * blockDim.x + threadIdx.x;
  if (t >= n4) return;
  int i = t >> 6;
  int c = (t & 63) << 2;
  float4 v = *(const float4*)(AGG + (size_t)i * HID + c);
  float4 bb = *(const float4*)(b + c);
  v4h o;
  o[0] = (_Float16)fmaxf(v.x + bb.x, 0.0f);
  o[1] = (_Float16)fmaxf(v.y + bb.y, 0.0f);
  o[2] = (_Float16)fmaxf(v.z + bb.z, 0.0f);
  o[3] = (_Float16)fmaxf(v.w + bb.w, 0.0f);
  *(v4h*)(out + (size_t)i * HID + c) = o;
}

// pool: pooled[g,:] += relu(AGG[i,:]+b5); cnt[g] += 1
__global__ void k_pool(const float* __restrict__ AGG, const float* __restrict__ b,
                       const int* __restrict__ batch, float* __restrict__ pooled,
                       float* __restrict__ cnt, int n4) {
  int t = blockIdx.x * blockDim.x + threadIdx.x;
  if (t >= n4) return;
  int i = t >> 6;
  int c = (t & 63) << 2;
  int g = batch[i];
  float4 v = *(const float4*)(AGG + (size_t)i * HID + c);
  float4 bb = *(const float4*)(b + c);
  float* o = pooled + (size_t)g * HID + c;
  atomAddF(o + 0, fmaxf(v.x + bb.x, 0.0f));
  atomAddF(o + 1, fmaxf(v.y + bb.y, 0.0f));
  atomAddF(o + 2, fmaxf(v.z + bb.z, 0.0f));
  atomAddF(o + 3, fmaxf(v.w + bb.w, 0.0f));
  if (c == 0) atomAddF(&cnt[g], 1.0f);
}

// out[g,c] = (pooled[g,:]/cnt[g]) @ Wfc + bfc
__global__ void k_fc(const float* __restrict__ pooled, const float* __restrict__ cnt,
                     const float* __restrict__ Wfc, const float* __restrict__ bfc,
                     float* __restrict__ out) {
  int g = blockIdx.x;
  int c = threadIdx.x;
  if (c >= NCLS) return;
  float s = 0.0f;
  for (int k = 0; k < HID; ++k)
    s += pooled[(size_t)g * HID + k] * Wfc[k * NCLS + c];
  float ic = 1.0f / fmaxf(cnt[g], 1.0f);
  out[g * NCLS + c] = s * ic + bfc[c];
}

// ---------------- host launcher ----------------

extern "C" void kernel_launch(void* const* d_in, const int* in_sizes, int n_in,
                              void* d_out, int out_size, void* d_ws, size_t ws_size,
                              hipStream_t stream) {
  (void)n_in; (void)out_size; (void)ws_size;
  const float* x     = (const float*)d_in[0];
  const int*   eidx  = (const int*)d_in[1];
  const int*   batch = (const int*)d_in[2];
  const float* W[5]  = {(const float*)d_in[3], (const float*)d_in[5], (const float*)d_in[7],
                        (const float*)d_in[9], (const float*)d_in[11]};
  const float* b[5]  = {(const float*)d_in[4], (const float*)d_in[6], (const float*)d_in[8],
                        (const float*)d_in[10], (const float*)d_in[12]};
  const float* Wfc = (const float*)d_in[13];
  const float* bfc = (const float*)d_in[14];

  const int N = in_sizes[0] / F_IN;      // 50000 (divisible by 16)
  const int E = in_sizes[1] / 2;         // 800000
  const int* src = eidx;
  const int* dst = eidx + E;

  uint8_t* wsp = (uint8_t*)d_ws;
  auto alloc = [&](size_t bytes) -> void* {
    void* p = wsp;
    wsp += (bytes + 255) & ~(size_t)255;
    return p;
  };
  float*     dinv   = (float*)alloc((size_t)N * 4);
  _Float16*  Xf16   = (_Float16*)alloc((size_t)N * F_IN * 2);
  _Float16*  F16A   = (_Float16*)alloc((size_t)N * HID * 2);
  float*     H      = (float*)alloc((size_t)N * HID * 4);
  float*     AGG    = (float*)alloc((size_t)N * HID * 4);
  _Float16*  Wt     = (_Float16*)alloc((size_t)HID * HID * 2);
  float*     pooled = (float*)alloc((size_t)NGRAPH * HID * 4);
  float*     cnt    = (float*)alloc((size_t)NGRAPH * 4);

  const int TB = 256;
  // degrees (self-loop folded in as init value 1.0) -> dinv
  k_fill  <<<(N + TB - 1) / TB, TB, 0, stream>>>(dinv, 1.0f, N);
  k_deg   <<<(E + TB - 1) / TB, TB, 0, stream>>>(dst, dinv, E);
  k_rsqrt <<<(N + TB - 1) / TB, TB, 0, stream>>>(dinv, N);
  // x -> f16
  k_f32tof16<<<((N * F_IN) + TB - 1) / TB, TB, 0, stream>>>(x, Xf16, N * F_IN);
  // zero pooled + counts
  k_fill<<<(NGRAPH * HID + TB - 1) / TB, TB, 0, stream>>>(pooled, 0.0f, NGRAPH * HID);
  k_fill<<<(NGRAPH + TB - 1) / TB, TB, 0, stream>>>(cnt, 0.0f, NGRAPH);

  const int n4 = N * (HID / 4);                       // node-feature chunks
  const long long etot = (long long)E * 64;           // edge-feature chunks

  for (int L = 0; L < 5; ++L) {
    const int K = (L == 0) ? F_IN : HID;
    k_wprep<<<(K * HID + TB - 1) / TB, TB, 0, stream>>>(W[L], Wt, K);
    if (L == 0)
      k_gemm_wmma<F_IN><<<N / 16, 128, 0, stream>>>(Xf16, Wt, H);
    else
      k_gemm_wmma<HID><<<N / 16, 128, 0, stream>>>(F16A, Wt, H);
    k_selfinit<<<(n4 + TB - 1) / TB, TB, 0, stream>>>(H, dinv, AGG, n4);
    k_edgeagg<<<(int)((etot + TB - 1) / TB), TB, 0, stream>>>(src, dst, dinv, H, AGG, E);
    if (L < 4)
      k_finalize<<<(n4 + TB - 1) / TB, TB, 0, stream>>>(AGG, b[L], F16A, n4);
    else
      k_pool<<<(n4 + TB - 1) / TB, TB, 0, stream>>>(AGG, b[4], batch, pooled, cnt, n4);
  }

  k_fc<<<NGRAPH, 32, 0, stream>>>(pooled, cnt, Wfc, bfc, (float*)d_out);
}
